// RNN_67654324847424
// MI455X (gfx1250) — matrix operand order repaired and checked by reading, not compiled, not run on hardware
//
#include <hip/hip_runtime.h>

// Problem dims (fixed by the reference).
#define T_STEPS 512
#define BATCH   256
#define I_DIM   256
#define H_DIM   512
#define L_DIM   1024
#define O_DIM   256

typedef __attribute__((ext_vector_type(16))) __bf16 v16bf;
typedef __attribute__((ext_vector_type(8)))  __bf16 v8bf;
typedef __attribute__((ext_vector_type(4)))  __bf16 v4bf;
typedef __attribute__((ext_vector_type(8)))  float  v8f;

union frag16 { v16bf v; struct { v8bf lo, hi; } h; };

// ---------------------------------------------------------------------------
// WMMA fragment loaders (wave32, v_wmma_f32_16x16x32_bf16), ISA 7.12.2 layout.
//   A (16x32 MxK, row-major, lda = K):
//     lanes 0-15: row M=lane, K={0..7,16..23}; lanes 16-31: row M=lane-16,
//     K={8..15,24..31}.  -> per lane: two contiguous 16B loads at +kh, +16+kh.
//   B (32x16 KxN) from TRANSPOSED weights Wt[N][K]:
//     lanes 0-15: col N=lane, K={0..7,16..23}; lanes 16-31: col N=lane-16,
//     K={8..15,24..31}.  -> per lane: two contiguous 16B loads at +kh, +16+kh.
//   C/D (16x16 f32): VGPR r -> row r + 8*(lane>=16), col = lane&15.
// ---------------------------------------------------------------------------
__device__ __forceinline__ const __bf16* a_lane_base(const __bf16* A, int lda) {
    const int lane = threadIdx.x & 31;
    return A + (lane & 15) * lda + ((lane >> 4) << 3);   // +0 or +8
}
__device__ __forceinline__ const __bf16* b_lane_base(const __bf16* Wt, int ldk) {
    const int lane = threadIdx.x & 31;
    return Wt + (size_t)(lane & 15) * ldk + ((lane >> 4) << 3);  // +0 or +8
}
__device__ __forceinline__ v16bf load_a(const __bf16* pa) {
    frag16 f;
    f.h.lo = *(const v8bf*)(pa);
    f.h.hi = *(const v8bf*)(pa + 16);
    return f.v;
}
__device__ __forceinline__ v16bf load_b(const __bf16* pb) {
    frag16 f;
    f.h.lo = *(const v8bf*)(pb);
    f.h.hi = *(const v8bf*)(pb + 16);
    return f.v;
}
__device__ __forceinline__ v8f wmma_bf16(v16bf a, v16bf b, v8f c) {
    return __builtin_amdgcn_wmma_f32_16x16x32_bf16(false, a, false, b,
                                                   (short)0, c, false, false);
}

// ---------------------------------------------------------------------------
// Tiled GEMM kernels for the time-parallel phases.
// Block = 256 threads = 8 waves; block stages 16 rows of A into LDS (bf16);
// each wave owns TWO adjacent 16-wide N tiles (shared A frag, 2 WMMA chains).
// Grid = (M/16, N/256).  Weights passed TRANSPOSED: Wt[N][K].
// ---------------------------------------------------------------------------

// C_bf16[M,N] = relu(A_f32[M,K] @ W[K,N] + bias[N])
__global__ __launch_bounds__(256) void gemm_relu_bf16(
    const float* __restrict__ A, const __bf16* __restrict__ Wt,
    const float* __restrict__ bias, __bf16* __restrict__ C,
    int N, int K)
{
    __shared__ __attribute__((aligned(32))) __bf16 sA[16 * 1024];
    const int m0   = blockIdx.x * 16;
    const int wave = threadIdx.x >> 5;
    const int n0   = (blockIdx.y * 8 + wave) * 32;

    // Stage A rows to LDS: float4 global reads -> packed v4bf LDS writes.
    for (int idx = threadIdx.x * 4; idx < 16 * K; idx += 256 * 4) {
        int r = idx / K, c = idx - r * K;
        const float4 f = *(const float4*)&A[(size_t)(m0 + r) * K + c];
        v4bf p; p[0] = (__bf16)f.x; p[1] = (__bf16)f.y; p[2] = (__bf16)f.z; p[3] = (__bf16)f.w;
        *(v4bf*)&sA[idx] = p;
    }
    __syncthreads();

    const __bf16* pa  = a_lane_base(sA, K);
    const __bf16* pb0 = b_lane_base(Wt + (size_t)(n0 +  0) * K, K);
    const __bf16* pb1 = b_lane_base(Wt + (size_t)(n0 + 16) * K, K);
    v8f acc0 = {}, acc1 = {};
#pragma unroll 2
    for (int k0 = 0; k0 < K; k0 += 32) {
        v16bf a = load_a(pa);
        acc0 = wmma_bf16(a, load_b(pb0), acc0);
        acc1 = wmma_bf16(a, load_b(pb1), acc1);
        pa += 32; pb0 += 32; pb1 += 32;
    }
    const int lane = threadIdx.x & 31;
    const int cn = lane & 15;
    const int cm = (lane >> 4) << 3;
    const float bv0 = bias[n0 + cn], bv1 = bias[n0 + 16 + cn];
#pragma unroll
    for (int r = 0; r < 8; ++r) {
        size_t row = (size_t)(m0 + cm + r) * N;
        C[row + n0 + cn]      = (__bf16)fmaxf(acc0[r] + bv0, 0.0f);
        C[row + n0 + 16 + cn] = (__bf16)fmaxf(acc1[r] + bv1, 0.0f);
    }
}

// C_f32[M,N] = A_bf16[M,K] @ W[K,N] + bias[N]
__global__ __launch_bounds__(256) void gemm_bias_f32(
    const __bf16* __restrict__ A, const __bf16* __restrict__ Wt,
    const float* __restrict__ bias, float* __restrict__ C,
    int N, int K)
{
    __shared__ __attribute__((aligned(32))) __bf16 sA[16 * 1024];
    const int m0   = blockIdx.x * 16;
    const int wave = threadIdx.x >> 5;
    const int n0   = (blockIdx.y * 8 + wave) * 32;

    for (int idx = threadIdx.x * 8; idx < 16 * K; idx += 256 * 8) {
        int r = idx / K, c = idx - r * K;
        *(v8bf*)&sA[idx] = *(const v8bf*)&A[(size_t)(m0 + r) * K + c];
    }
    __syncthreads();

    const __bf16* pa  = a_lane_base(sA, K);
    const __bf16* pb0 = b_lane_base(Wt + (size_t)(n0 +  0) * K, K);
    const __bf16* pb1 = b_lane_base(Wt + (size_t)(n0 + 16) * K, K);
    v8f acc0 = {}, acc1 = {};
#pragma unroll 2
    for (int k0 = 0; k0 < K; k0 += 32) {
        v16bf a = load_a(pa);
        acc0 = wmma_bf16(a, load_b(pb0), acc0);
        acc1 = wmma_bf16(a, load_b(pb1), acc1);
        pa += 32; pb0 += 32; pb1 += 32;
    }
    const int lane = threadIdx.x & 31;
    const int cn = lane & 15;
    const int cm = (lane >> 4) << 3;
    const float bv0 = bias[n0 + cn], bv1 = bias[n0 + 16 + cn];
#pragma unroll
    for (int r = 0; r < 8; ++r) {
        size_t row = (size_t)(m0 + cm + r) * N;
        C[row + n0 + cn]      = acc0[r] + bv0;
        C[row + n0 + 16 + cn] = acc1[r] + bv1;
    }
}

// ---------------------------------------------------------------------------
// Persistent recurrent kernel: internal t-loop; grid barrier between stages.
// All recurrent operands (h 256KB, act 512KB, wh1/wh2 bf16 2MB) are
// L2-resident; pre[t] streams 512KB/step from HBM.
// ---------------------------------------------------------------------------
#define RNN_BLOCKS 64

__device__ __forceinline__ void grid_barrier(int* cnt, int* gen, int nblocks) {
    __threadfence();
    __syncthreads();
    if (threadIdx.x == 0) {
        volatile int* vgen = gen;
        int g = *vgen;
        if (atomicAdd(cnt, 1) == nblocks - 1) {
            *cnt = 0;
            __threadfence();
            atomicAdd(gen, 1);
        } else {
            while (*vgen == g) { __builtin_amdgcn_s_sleep(2); }
        }
    }
    __syncthreads();
}

__global__ __launch_bounds__(256) void rnn_recurrent(
    const __bf16* __restrict__ wh1t,      // [L,H] transposed (bf16)
    const float* __restrict__ bh1,
    const __bf16* __restrict__ wh2t,      // [H,L] transposed (bf16)
    const float* __restrict__ bh2,
    const float* __restrict__ pre,        // [T,B,H] precomputed input MLP
    float* __restrict__ hf,               // [B,H] f32 state (for output MLP)
    __bf16* __restrict__ hb,              // [B,H] bf16 state (WMMA operand)
    __bf16* __restrict__ act,             // [B,L] scratch
    int* cnt, int* gen)
{
    const int nwaves = RNN_BLOCKS * 8;    // 512 waves
    const int gwave  = blockIdx.x * 8 + (threadIdx.x >> 5);
    const int lane = threadIdx.x & 31;
    const int cn = lane & 15;
    const int cm = (lane >> 4) << 3;

    for (int t = 0; t < T_STEPS; ++t) {
        // Stage 1: act = relu(hb @ wh1 + bh1).  M=256,K=512,N=1024.
        // 512 tile-PAIRS (two N-tiles/wave) == 512 waves: one pair each.
        {
            const int pairidx = gwave;                    // 0..511
            const int m0 = (pairidx & 15) * 16;
            const int n0 = (pairidx >> 4) * 32;
            const __bf16* pa  = a_lane_base(hb + m0 * H_DIM, H_DIM);
            const __bf16* pb0 = b_lane_base(wh1t + (size_t)(n0 +  0) * H_DIM, H_DIM);
            const __bf16* pb1 = b_lane_base(wh1t + (size_t)(n0 + 16) * H_DIM, H_DIM);
            v8f acc0 = {}, acc1 = {};
#pragma unroll 2
            for (int k0 = 0; k0 < H_DIM; k0 += 32) {
                v16bf a = load_a(pa);
                acc0 = wmma_bf16(a, load_b(pb0), acc0);
                acc1 = wmma_bf16(a, load_b(pb1), acc1);
                pa += 32; pb0 += 32; pb1 += 32;
            }
            const float bv0 = bh1[n0 + cn], bv1 = bh1[n0 + 16 + cn];
#pragma unroll
            for (int r = 0; r < 8; ++r) {
                int row = (m0 + cm + r) * L_DIM;
                act[row + n0 + cn]      = (__bf16)fmaxf(acc0[r] + bv0, 0.0f);
                act[row + n0 + 16 + cn] = (__bf16)fmaxf(acc1[r] + bv1, 0.0f);
            }
        }
        grid_barrier(cnt, gen, RNN_BLOCKS);

        // Stage 2: h = tanh(pre[t] + act @ wh2 + bh2).  M=256,K=1024,N=512.
        // 512 single tiles == 512 waves.
        {
            const float* pre_t = pre + (size_t)t * BATCH * H_DIM;
            const int tile = gwave;                       // 0..511
            const int m0 = (tile & 15) * 16;
            const int n0 = (tile >> 4) * 16;
            const __bf16* pa = a_lane_base(act + m0 * L_DIM, L_DIM);
            const __bf16* pb = b_lane_base(wh2t + (size_t)n0 * L_DIM, L_DIM);
            v8f acc = {};
#pragma unroll 2
            for (int k0 = 0; k0 < L_DIM; k0 += 32) {
                acc = wmma_bf16(load_a(pa), load_b(pb), acc);
                pa += 32; pb += 32;
            }
            const float bv = bh2[n0 + cn];
#pragma unroll
            for (int r = 0; r < 8; ++r) {
                const int row = m0 + cm + r, col = n0 + cn;
                float v = tanhf(acc[r] + bv + pre_t[row * H_DIM + col]);
                hf[row * H_DIM + col] = v;
                hb[row * H_DIM + col] = (__bf16)v;
            }
        }
        grid_barrier(cnt, gen, RNN_BLOCKS);
    }
}

// ---------------------------------------------------------------------------
// Helpers: transposing weight conversion + state/barrier init.
// Wt[n*K + k] = (bf16) W[k*N + n]   (runs once; weights are tiny).
// ---------------------------------------------------------------------------
__global__ void f32_to_bf16_t(const float* __restrict__ W, __bf16* __restrict__ Wt,
                              int K, int N) {
    int i = blockIdx.x * blockDim.x + threadIdx.x;
    if (i < K * N) {
        int k = i / N, n = i - k * N;
        Wt[(size_t)n * K + k] = (__bf16)W[i];
    }
}

__global__ void init_state(float* hf, __bf16* hb, int n, int* cnt, int* gen) {
    int i = blockIdx.x * blockDim.x + threadIdx.x;
    if (i < n) { hf[i] = 0.0f; hb[i] = (__bf16)0.0f; }
    if (i == 0) { *cnt = 0; *gen = 0; }
}

// ---------------------------------------------------------------------------
// Host-side orchestration. Workspace ~543 MB:
//   pre f32 [T*B*H] 256MB | act bf16 [T*B*L] 256MB | 6 transposed bf16
//   weights ~5.25MB | hf f32 + hb bf16 + barrier counters.
// ---------------------------------------------------------------------------
extern "C" void kernel_launch(void* const* d_in, const int* in_sizes, int n_in,
                              void* d_out, int out_size, void* d_ws, size_t ws_size,
                              hipStream_t stream) {
    const float* x   = (const float*)d_in[0];
    const float* wi1 = (const float*)d_in[1];
    const float* bi1 = (const float*)d_in[2];
    const float* wi2 = (const float*)d_in[3];
    const float* bi2 = (const float*)d_in[4];
    const float* wh1 = (const float*)d_in[5];
    const float* bh1 = (const float*)d_in[6];
    const float* wh2 = (const float*)d_in[7];
    const float* bh2 = (const float*)d_in[8];
    const float* wo1 = (const float*)d_in[9];
    const float* bo1 = (const float*)d_in[10];
    const float* wo2 = (const float*)d_in[11];
    const float* bo2 = (const float*)d_in[12];

    char* ws = (char*)d_ws;
    size_t off = 0;
    float*  pre  = (float*)(ws + off);  off += (size_t)T_STEPS * BATCH * H_DIM * 4;
    __bf16* act  = (__bf16*)(ws + off); off += (size_t)T_STEPS * BATCH * L_DIM * 2;
    __bf16* wi1t = (__bf16*)(ws + off); off += (size_t)I_DIM * L_DIM * 2;
    __bf16* wi2t = (__bf16*)(ws + off); off += (size_t)L_DIM * H_DIM * 2;
    __bf16* wh1t = (__bf16*)(ws + off); off += (size_t)H_DIM * L_DIM * 2;
    __bf16* wh2t = (__bf16*)(ws + off); off += (size_t)L_DIM * H_DIM * 2;
    __bf16* wo1t = (__bf16*)(ws + off); off += (size_t)H_DIM * L_DIM * 2;
    __bf16* wo2t = (__bf16*)(ws + off); off += (size_t)L_DIM * O_DIM * 2;
    float*  hf   = (float*)(ws + off);  off += (size_t)BATCH * H_DIM * 4;
    __bf16* hb   = (__bf16*)(ws + off); off += (size_t)BATCH * H_DIM * 2;
    int*    cnt  = (int*)(ws + off);    off += 128;
    int*    gen  = (int*)(ws + off);    off += 128;

    const int TB = T_STEPS * BATCH;   // 131072 rows

    // Weights -> transposed bf16 (L2-resident for the whole run).
    auto cvt = [&](const float* s, __bf16* d, int K, int N) {
        f32_to_bf16_t<<<(K * N + 255) / 256, 256, 0, stream>>>(s, d, K, N);
    };
    cvt(wi1, wi1t, I_DIM, L_DIM);
    cvt(wi2, wi2t, L_DIM, H_DIM);
    cvt(wh1, wh1t, H_DIM, L_DIM);
    cvt(wh2, wh2t, L_DIM, H_DIM);
    cvt(wo1, wo1t, H_DIM, L_DIM);
    cvt(wo2, wo2t, L_DIM, O_DIM);
    init_state<<<(BATCH * H_DIM + 255) / 256, 256, 0, stream>>>(hf, hb, BATCH * H_DIM, cnt, gen);

    // Phase 1 (time-parallel): pre[t] = relu(x @ wi1 + bi1) @ wi2 + bi2
    gemm_relu_bf16<<<dim3(TB / 16, L_DIM / 256), 256, 0, stream>>>(x, wi1t, bi1, act, L_DIM, I_DIM);
    gemm_bias_f32 <<<dim3(TB / 16, H_DIM / 256), 256, 0, stream>>>(act, wi2t, bi2, pre, H_DIM, L_DIM);

    // Phase 2 (sequential): persistent kernel, internal t-loop + grid barriers.
    rnn_recurrent<<<RNN_BLOCKS, 256, 0, stream>>>(wh1t, bh1, wh2t, bh2, pre, hf, hb, act, cnt, gen);

    // Phase 3: out = relu(h @ wo1 + bo1) @ wo2 + bo2
    gemm_relu_bf16<<<dim3(BATCH / 16, L_DIM / 256), 256, 0, stream>>>(hf, wo1t, bo1, act, L_DIM, H_DIM);
    gemm_bias_f32 <<<dim3(BATCH / 16, O_DIM / 256), 256, 0, stream>>>(act, wo2t, bo2, (float*)d_out, O_DIM, L_DIM);
}
